// GTG_Module_77498389889235
// MI455X (gfx1250) — matrix operand order repaired and checked by reading, not compiled, not run on hardware
//
#include <hip/hip_runtime.h>
#include <hip/hip_bf16.h>
#include <math.h>

typedef __attribute__((ext_vector_type(16))) _Float16 v16h;
typedef __attribute__((ext_vector_type(8)))  float    v8f;

#define BQ 512          // batch
#define NC 10
#define N_LAB 256
#define MAX_ITER 30
#define GTG_TOL 0.001f

// ---------------------------------------------------------------------------
// WMMA 16x16x32 f16 fragment layouts (wave32, ISA 7.12.2):
//   A 16x32 (MxK): lanes0-15 -> M=lane, VGPR v holds K = (v<4 ? 2v : 8+2v)+8*hi
//   B 32x16 (KxN): lane%16 -> N, VGPR v holds K = (v<4 ? 2v : 8+2v)+8*hi, same order as A
//   C/D 16x16 f32: VGPR r -> M = r + 8*hi, lane%16 -> N
// Block tile 64x64, 256 threads = 8 waves; wave w owns m-sub (w&3)*16 and
// n-subs (w>>2)*32 + {0,16}.
// ---------------------------------------------------------------------------

template <int C, int S>
__global__ void __launch_bounds__(256)
conv_wmma_kernel(const float* __restrict__ x, const float* __restrict__ wgt,
                 const float* __restrict__ bias, float* __restrict__ y,
                 float* __restrict__ chSum, float* __restrict__ chSq) {
  constexpr int S2 = S * S;
  constexpr int K = 9 * C;
  const int tid = threadIdx.x;
  const int wave = tid >> 5;
  const int lane = tid & 31;
  const int l16 = lane & 15;
  const int hi = lane >> 4;

  const int nBase = blockIdx.x * 64;   // output position base (b*S*S + pix)
  const int mBase = blockIdx.y * 64;   // output channel base

  __shared__ _Float16 As[64][40];      // [m][k]
  __shared__ _Float16 Bs[32][72];      // [k][n]
  __shared__ float sSum[64];
  __shared__ float sSq[64];
  if (tid < 64) { sSum[tid] = 0.f; sSq[tid] = 0.f; }

  const int msub = (wave & 3) * 16;
  const int nsub = (wave >> 2) * 32;

  v8f acc0 = {};
  v8f acc1 = {};

  for (int k0 = 0; k0 < K; k0 += 32) {
    __syncthreads();
    // A tile: weights, As[m][kk] = W[co=mBase+m][ci][kh][kw], k = ci*9 + r
    for (int e = tid; e < 64 * 32; e += 256) {
      int m  = e >> 5;
      int kk = e & 31;
      int k  = k0 + kk;
      int ci = k / 9;              // compile-time-strength-reduced (K known)
      int r  = k - ci * 9;
      As[m][kk] = (_Float16)wgt[((mBase + m) * C + ci) * 9 + r];
    }
    // B tile: on-the-fly im2col with SAME zero padding (all /,% are shifts now)
    for (int e = tid; e < 32 * 64; e += 256) {
      int kk = e >> 6;
      int nn = e & 63;
      int n  = nBase + nn;
      int k  = k0 + kk;
      int ci = k / 9;
      int r  = k - ci * 9;
      int kh = r / 3;
      int kw = r - kh * 3;
      int img = n / S2;
      int pix = n & (S2 - 1);
      int h = pix / S;
      int w = pix & (S - 1);
      int ih = h + kh - 1;
      int iw = w + kw - 1;
      float v = 0.f;
      if ((unsigned)ih < (unsigned)S && (unsigned)iw < (unsigned)S)
        v = x[(img * C + ci) * S2 + ih * S + iw];
      Bs[kk][nn] = (_Float16)v;
    }
    __syncthreads();
    if (k0 + 32 < K)  // pull next weight slab toward L2 (global_prefetch_b8)
      __builtin_prefetch(&wgt[(mBase * C + (k0 + 32) / 9) * 9], 0, 3);

    v16h a, b0, b1;
#pragma unroll
    for (int e = 0; e < 16; ++e) {
      int v  = e >> 1;
      int kA = ((v < 4) ? (2 * v) : (8 + 2 * v)) + hi * 8 + (e & 1);
      a[e]  = As[msub + l16][kA];
    int kB = (e & 7) + hi * 8 + (e >> 3) * 16;
      b0[e] = Bs[kB][nsub + l16];
      b1[e] = Bs[kB][nsub + 16 + l16];
    }
    acc0 = __builtin_amdgcn_wmma_f32_16x16x32_f16(false, a, false, b0, (short)0, acc0, false, false);
    acc1 = __builtin_amdgcn_wmma_f32_16x16x32_f16(false, a, false, b1, (short)0, acc1, false, false);
  }
  __syncthreads();

  // epilogue: +bias, store y, accumulate per-channel sum / sumsq for BN
#pragma unroll
  for (int r = 0; r < 8; ++r) {
    int m  = msub + r + hi * 8;
    int co = mBase + m;
    float bv = bias[co];
    {
      int n   = nBase + nsub + l16;
      int img = n / S2;
      int pix = n & (S2 - 1);
      float v = acc0[r] + bv;
      y[(img * C + co) * S2 + pix] = v;
      atomicAdd(&sSum[m], v);
      atomicAdd(&sSq[m], v * v);
    }
    {
      int n   = nBase + nsub + 16 + l16;
      int img = n / S2;
      int pix = n & (S2 - 1);
      float v = acc1[r] + bv;
      y[(img * C + co) * S2 + pix] = v;
      atomicAdd(&sSum[m], v);
      atomicAdd(&sSq[m], v * v);
    }
  }
  __syncthreads();
  if (tid < 64) {
    atomicAdd(&chSum[mBase + tid], sSum[tid]);
    atomicAdd(&chSq[mBase + tid], sSq[tid]);
  }
}

template <int C, int S>
__global__ void __launch_bounds__(128)
bn_pool_kernel(const float* __restrict__ y, const float* __restrict__ chSum,
               const float* __restrict__ chSq, const float* __restrict__ g,
               const float* __restrict__ beta, float* __restrict__ pooled) {
  constexpr int S2 = S * S;
  const int b = blockIdx.x;
  const int c = blockIdx.y;
  constexpr float inv = 1.0f / (512.0f * (float)S2);
  float mean  = chSum[c] * inv;
  float var   = chSq[c] * inv - mean * mean;
  float scale = g[c] * rsqrtf(var + 1e-5f);
  float shift = beta[c] - scale * mean;
  const float* yp = y + (b * C + c) * S2;
  float s = 0.f;
  for (int i = threadIdx.x; i < S2; i += 128)
    s += fmaxf(scale * yp[i] + shift, 0.f);
  __shared__ float red[128];
  red[threadIdx.x] = s;
  __syncthreads();
  for (int off = 64; off > 0; off >>= 1) {
    if (threadIdx.x < off) red[threadIdx.x] += red[threadIdx.x + off];
    __syncthreads();
  }
  if (threadIdx.x == 0) pooled[b * C + c] = red[0] * (1.0f / (float)S2);
}

// linear: o = relu(pooled(512xC) @ lw^T(Cx128) + lb), written to ocat col slice
template <int C>
__global__ void __launch_bounds__(256)
linear_wmma_kernel(const float* __restrict__ pooled, const float* __restrict__ lw,
                   const float* __restrict__ lb, float* __restrict__ ocat,
                   int colOff) {
  const int tid = threadIdx.x;
  const int wave = tid >> 5;
  const int lane = tid & 31;
  const int l16 = lane & 15;
  const int hi = lane >> 4;
  const int nBase = blockIdx.x * 64;   // feature base (0 or 64)
  const int mBase = blockIdx.y * 64;   // batch base
  __shared__ _Float16 As[64][40];
  __shared__ _Float16 Bs[32][72];
  const int msub = (wave & 3) * 16;
  const int nsub = (wave >> 2) * 32;
  v8f acc0 = {}, acc1 = {};
  for (int k0 = 0; k0 < C; k0 += 32) {
    __syncthreads();
    for (int e = tid; e < 64 * 32; e += 256) {
      int m = e >> 5, kk = e & 31;
      As[m][kk] = (_Float16)pooled[(mBase + m) * C + k0 + kk];
    }
    for (int e = tid; e < 32 * 64; e += 256) {
      int kk = e >> 6, nn = e & 63;
      Bs[kk][nn] = (_Float16)lw[(nBase + nn) * C + k0 + kk];
    }
    __syncthreads();
    v16h a, b0, b1;
#pragma unroll
    for (int e = 0; e < 16; ++e) {
      int v  = e >> 1;
      int kA = ((v < 4) ? (2 * v) : (8 + 2 * v)) + hi * 8 + (e & 1);
      a[e]  = As[msub + l16][kA];
    int kB = (e & 7) + hi * 8 + (e >> 3) * 16;
      b0[e] = Bs[kB][nsub + l16];
      b1[e] = Bs[kB][nsub + 16 + l16];
    }
    acc0 = __builtin_amdgcn_wmma_f32_16x16x32_f16(false, a, false, b0, (short)0, acc0, false, false);
    acc1 = __builtin_amdgcn_wmma_f32_16x16x32_f16(false, a, false, b1, (short)0, acc1, false, false);
  }
#pragma unroll
  for (int r = 0; r < 8; ++r) {
    int m = mBase + msub + r + hi * 8;
    {
      int n = nBase + nsub + l16;
      ocat[m * 512 + colOff + n] = fmaxf(acc0[r] + lb[n], 0.f);
    }
    {
      int n = nBase + nsub + 16 + l16;
      ocat[m * 512 + colOff + n] = fmaxf(acc1[r] + lb[n], 0.f);
    }
  }
}

__global__ void __launch_bounds__(32)
ypred_kernel(const float* __restrict__ ocat, const float* __restrict__ fw,
             const float* __restrict__ fb, float* __restrict__ out) {
  int b = blockIdx.x;
  float s = 0.f;
  for (int k = threadIdx.x; k < 512; k += 32) s += ocat[b * 512 + k] * fw[k];
  for (int off = 16; off > 0; off >>= 1) s += __shfl_down(s, off, 32);
  if (threadIdx.x == 0) out[b] = s + fb[0];
}

// ------------------------------- GTG --------------------------------------

__global__ void __launch_bounds__(256)
e2f16_kernel(const float* __restrict__ E, _Float16* __restrict__ Ef16) {
  int t = blockIdx.x * 256 + threadIdx.x;   // 1024 blocks x 256 = 262144
  Ef16[t] = (_Float16)E[t];
}

__global__ void __launch_bounds__(128)
sq_kernel(const float* __restrict__ E, float* __restrict__ sq) {
  int row = blockIdx.x;
  float s = 0.f;
  for (int k = threadIdx.x; k < 512; k += 128) {
    float v = E[row * 512 + k];
    s += v * v;
  }
  __shared__ float red[128];
  red[threadIdx.x] = s;
  __syncthreads();
  for (int off = 64; off > 0; off >>= 1) {
    if (threadIdx.x < off) red[threadIdx.x] += red[threadIdx.x + off];
    __syncthreads();
  }
  if (threadIdx.x == 0) sq[row] = red[0];
}

// A0 = clip(sqrt(max(sq_i + sq_j - 2*E E^T, 0)), 0.1, 1.0), diag forced 0.
// Tiles are staged LDS via ASYNCcnt-tracked global_load_async_to_lds_b128:
// both A (rows of E) and B (rows of E, kept row-major in LDS -> "transposed"
// B tile) are contiguous 16B chunks; 256 threads x 1 chunk = one 4KB tile.
__global__ void __launch_bounds__(256)
gram_wmma_kernel(const _Float16* __restrict__ Ef16, const float* __restrict__ sq,
                 float* __restrict__ A0) {
  const int tid = threadIdx.x;
  const int wave = tid >> 5;
  const int lane = tid & 31;
  const int l16 = lane & 15;
  const int hi = lane >> 4;
  const int nBase = blockIdx.x * 64;
  const int mBase = blockIdx.y * 64;
  __shared__ _Float16 As[64][32];     // [m][k], row-contiguous
  __shared__ _Float16 Bst[64][32];    // [n][k], row-contiguous (B transposed)
  const int msub = (wave & 3) * 16;
  const int nsub = (wave >> 2) * 32;
  const int row = tid >> 2;           // 0..63
  const int chunk = (tid & 3) * 8;    // f16 offset of 16B chunk
  v8f acc0 = {}, acc1 = {};
  for (int k0 = 0; k0 < 512; k0 += 32) {
    __syncthreads();                  // previous-iter readers done
    {
      const _Float16* gA = Ef16 + (mBase + row) * 512 + k0 + chunk;
      const _Float16* gB = Ef16 + (nBase + row) * 512 + k0 + chunk;
      unsigned lA = (unsigned)(size_t)&As[row][chunk];   // flat low 32b = LDS off
      unsigned lB = (unsigned)(size_t)&Bst[row][chunk];
      asm volatile("global_load_async_to_lds_b128 %0, %1, off"
                   :: "v"(lA), "v"(gA) : "memory");
      asm volatile("global_load_async_to_lds_b128 %0, %1, off"
                   :: "v"(lB), "v"(gB) : "memory");
      asm volatile("s_wait_asynccnt 0x0" ::: "memory");
    }
    __syncthreads();                  // all waves' async copies landed
    v16h a, b0, b1;
#pragma unroll
    for (int e = 0; e < 16; ++e) {
      int v  = e >> 1;
      int kA = ((v < 4) ? (2 * v) : (8 + 2 * v)) + hi * 8 + (e & 1);
      a[e]  = As[msub + l16][kA];
      int kB = (e & 7) + hi * 8 + (e >> 3) * 16;
      b0[e] = Bst[nsub + l16][kB];
      b1[e] = Bst[nsub + 16 + l16][kB];
    }
    acc0 = __builtin_amdgcn_wmma_f32_16x16x32_f16(false, a, false, b0, (short)0, acc0, false, false);
    acc1 = __builtin_amdgcn_wmma_f32_16x16x32_f16(false, a, false, b1, (short)0, acc1, false, false);
  }
#pragma unroll
  for (int r = 0; r < 8; ++r) {
    int mG = mBase + msub + r + hi * 8;
#pragma unroll
    for (int half = 0; half < 2; ++half) {
      int nG = nBase + nsub + half * 16 + l16;
      float g = half ? acc1[r] : acc0[r];
      float d2 = sq[mG] + sq[nG] - 2.f * g;
      if (mG == nG) d2 = 0.f;                    // exact ||e-e||^2
      float a0 = sqrtf(fmaxf(d2, 0.f));
      A0[mG * 512 + nG] = fminf(fmaxf(a0, 0.1f), 1.0f);
    }
  }
}

// stable 7th-smallest (matches stable argsort[:,6]) then sigma
__global__ void __launch_bounds__(32)
seventh_sig_kernel(const float* __restrict__ A0, const float* __restrict__ E,
                   float* __restrict__ sig) {
  const int row = blockIdx.x;
  const int lane = threadIdx.x;
  float vals[16];
#pragma unroll
  for (int j = 0; j < 16; ++j) vals[j] = A0[row * 512 + lane * 16 + j];
  int sel = 0;
  for (int it = 0; it < 7; ++it) {
    float mv = 3.4e38f;
    int mi = 0x7fffffff;
#pragma unroll
    for (int j = 0; j < 16; ++j) {
      if (vals[j] < mv) { mv = vals[j]; mi = lane * 16 + j; }
    }
    for (int off = 16; off > 0; off >>= 1) {
      float ov = __shfl_down(mv, off, 32);
      int   oi = __shfl_down(mi, off, 32);
      if (ov < mv || (ov == mv && oi < mi)) { mv = ov; mi = oi; }
    }
    mi = __shfl(mi, 0, 32);
    sel = mi;
    if ((mi >> 4) == lane) vals[mi & 15] = 3.4e38f;   // remove, keep stability
  }
  float s = 0.f;
  for (int j = lane; j < 512; j += 32) {
    float d = E[row * 512 + j] - E[sel * 512 + j];
    s += d * d;
  }
  for (int off = 16; off > 0; off >>= 1) s += __shfl_down(s, off, 32);
  if (lane == 0) sig[row] = fminf(fmaxf(sqrtf(s), 0.1f), 1.0f);
}

__global__ void __launch_bounds__(256)
a_build_kernel(const float* __restrict__ A0, const float* __restrict__ sig,
               float* __restrict__ A, float* __restrict__ Asum) {
  int idx = blockIdx.x * 256 + threadIdx.x;
  int i = idx >> 9, j = idx & 511;
  float a0 = A0[idx];
  float a = expf(-a0 * a0 / (sig[i] * sig[j]));
  a = fminf(fmaxf(a, 0.f), 1.f);
  A[idx] = a;
  __shared__ float red[256];
  red[threadIdx.x] = a;
  __syncthreads();
  for (int off = 128; off > 0; off >>= 1) {
    if (threadIdx.x < off) red[threadIdx.x] += red[threadIdx.x + off];
    __syncthreads();
  }
  if (threadIdx.x == 0) atomicAdd(Asum, red[0]);
}

__global__ void __launch_bounds__(256)
a_thresh_kernel(float* __restrict__ A, const float* __restrict__ Asum) {
  int idx = blockIdx.x * 256 + threadIdx.x;
  int i = idx >> 9, j = idx & 511;
  float mean = *Asum * (1.0f / 262144.0f);
  float a = A[idx];
  a = (a > mean) ? 1.0f : a;
  if (i < N_LAB && j < N_LAB) a = 1.0f - a;
  A[idx] = a;
}

__global__ void __launch_bounds__(256)
x_init_kernel(const int* __restrict__ labels, float* __restrict__ X,
              float* __restrict__ hist, float* __restrict__ diff2,
              int* __restrict__ flag) {
  int t = blockIdx.x * 256 + threadIdx.x;
  if (t < BQ * NC) {
    int i = t / NC, c = t - i * NC;
    X[t] = (i < N_LAB) ? ((labels[i] == c) ? 1.f : 0.f) : (1.0f / NC);
  }
  if (t < BQ * MAX_ITER) hist[t] = 0.f;
  if (t == 0) { *diff2 = 0.f; *flag = 0; }
}

__global__ void __launch_bounds__(256)
gtg_iter_a(const float* __restrict__ A, const float* __restrict__ X,
           float* __restrict__ Xn, float* __restrict__ hist,
           float* __restrict__ diff2, const int* __restrict__ flag, int iter) {
  if (*flag) return;
  const int row = blockIdx.x;
  const int tid = threadIdx.x;
  float acc[NC];
#pragma unroll
  for (int c = 0; c < NC; ++c) acc[c] = 0.f;
  for (int k = tid; k < 512; k += 256) {
    float a = A[row * 512 + k];
    const float* xr = X + k * NC;
#pragma unroll
    for (int c = 0; c < NC; ++c) acc[c] += a * xr[c];
  }
  __shared__ float red[NC * 256];
#pragma unroll
  for (int c = 0; c < NC; ++c) red[c * 256 + tid] = acc[c];
  __syncthreads();
  for (int off = 128; off > 0; off >>= 1) {
    if (tid < off)
#pragma unroll
      for (int c = 0; c < NC; ++c) red[c * 256 + tid] += red[c * 256 + tid + off];
    __syncthreads();
  }
  if (tid == 0) {
    float xn[NC];
    float s = 0.f;
#pragma unroll
    for (int c = 0; c < NC; ++c) {
      xn[c] = X[row * NC + c] * red[c * 256];
      s += xn[c];
    }
    float inv = 1.f / s;
    float ent = 0.f, d = 0.f;
#pragma unroll
    for (int c = 0; c < NC; ++c) {
      float v = xn[c] * inv;
      Xn[row * NC + c] = v;
      if (v > 0.f) ent -= v * logf(v);
      float dd = v - X[row * NC + c];
      d += dd * dd;
    }
    hist[row * MAX_ITER + iter] = ent;
    atomicAdd(diff2, d);
  }
}

__global__ void __launch_bounds__(256)
gtg_iter_b(float* __restrict__ X, const float* __restrict__ Xn,
           float* __restrict__ diff2, int* __restrict__ flag) {
  if (*flag) return;                 // single block: no cross-block race
  for (int t = threadIdx.x; t < BQ * NC; t += 256) X[t] = Xn[t];
  __syncthreads();
  if (threadIdx.x == 0) {
    float d = *diff2;
    if (sqrtf(d) <= GTG_TOL) *flag = 1;
    *diff2 = 0.f;
  }
}

__global__ void __launch_bounds__(256)
gtg_finalize(const float* __restrict__ hist, const float* __restrict__ X,
             float* __restrict__ out) {
  int t = blockIdx.x * 256 + threadIdx.x;
  if (t < BQ) {
    float s = 0.f;
    for (int j = 0; j < MAX_ITER; ++j) s += hist[t * MAX_ITER + j];
    out[512 + t] = s / (float)MAX_ITER;          // y_true
    out[6144 + t] = (t < N_LAB) ? 1.f : 0.f;     // labeled_mask
  }
  if (t < BQ * NC) out[1024 + t] = X[t];         // X
}

__global__ void zero_kernel(float* __restrict__ p, int n) {
  int t = blockIdx.x * 256 + threadIdx.x;
  if (t < n) p[t] = 0.f;
}

// ---------------------------------------------------------------------------

extern "C" void kernel_launch(void* const* d_in, const int* in_sizes, int n_in,
                              void* d_out, int out_size, void* d_ws, size_t ws_size,
                              hipStream_t stream) {
  (void)in_sizes; (void)n_in; (void)out_size; (void)ws_size;

  const float* fin_w  = (const float*)d_in[28];
  const float* fin_b  = (const float*)d_in[29];
  const float* E      = (const float*)d_in[30];
  const int*   labels = (const int*)d_in[31];
  float* out = (float*)d_out;

  char* ws = (char*)d_ws;
  size_t off = 0;
  auto alloc = [&](size_t bytes) -> void* {
    void* p = ws + off;
    off = (off + bytes + 255) & ~(size_t)255;
    return p;
  };
  float*     y       = (float*)alloc((size_t)512 * 64 * 32 * 32 * sizeof(float));
  float*     pooled  = (float*)alloc((size_t)512 * 512 * sizeof(float));
  float*     ocat    = (float*)alloc((size_t)512 * 512 * sizeof(float));
  float*     chStats = (float*)alloc(1024 * sizeof(float));   // [sum|sq]
  float*     A0      = (float*)alloc((size_t)512 * 512 * sizeof(float));
  float*     Amat    = (float*)alloc((size_t)512 * 512 * sizeof(float));
  _Float16*  Ef16    = (_Float16*)alloc((size_t)512 * 512 * sizeof(_Float16));
  float*     sq      = (float*)alloc(512 * sizeof(float));
  float*     sig     = (float*)alloc(512 * sizeof(float));
  float*     Asum    = (float*)alloc(sizeof(float));
  float*     X       = (float*)alloc(BQ * NC * sizeof(float));
  float*     Xn      = (float*)alloc(BQ * NC * sizeof(float));
  float*     hist    = (float*)alloc(BQ * MAX_ITER * sizeof(float));
  float*     diff2   = (float*)alloc(sizeof(float));
  int*       flag    = (int*)alloc(sizeof(int));

  // ----- branch pipeline (C,S compile-time per branch) -----
  for (int i = 0; i < 4; ++i) {
    const float* feat = (const float*)d_in[i * 7 + 0];
    const float* cw   = (const float*)d_in[i * 7 + 1];
    const float* cb   = (const float*)d_in[i * 7 + 2];
    const float* bg   = (const float*)d_in[i * 7 + 3];
    const float* bb   = (const float*)d_in[i * 7 + 4];
    const float* lw   = (const float*)d_in[i * 7 + 5];
    const float* lb   = (const float*)d_in[i * 7 + 6];

    zero_kernel<<<4, 256, 0, stream>>>(chStats, 1024);
    switch (i) {
      case 0: {
        dim3 gc(512 * 32 * 32 / 64, 64 / 64);
        conv_wmma_kernel<64, 32><<<gc, 256, 0, stream>>>(feat, cw, cb, y, chStats, chStats + 512);
        dim3 gp(512, 64);
        bn_pool_kernel<64, 32><<<gp, 128, 0, stream>>>(y, chStats, chStats + 512, bg, bb, pooled);
        dim3 gl(2, 8);
        linear_wmma_kernel<64><<<gl, 256, 0, stream>>>(pooled, lw, lb, ocat, 0);
        break;
      }
      case 1: {
        dim3 gc(512 * 16 * 16 / 64, 128 / 64);
        conv_wmma_kernel<128, 16><<<gc, 256, 0, stream>>>(feat, cw, cb, y, chStats, chStats + 512);
        dim3 gp(512, 128);
        bn_pool_kernel<128, 16><<<gp, 128, 0, stream>>>(y, chStats, chStats + 512, bg, bb, pooled);
        dim3 gl(2, 8);
        linear_wmma_kernel<128><<<gl, 256, 0, stream>>>(pooled, lw, lb, ocat, 128);
        break;
      }
      case 2: {
        dim3 gc(512 * 8 * 8 / 64, 256 / 64);
        conv_wmma_kernel<256, 8><<<gc, 256, 0, stream>>>(feat, cw, cb, y, chStats, chStats + 512);
        dim3 gp(512, 256);
        bn_pool_kernel<256, 8><<<gp, 128, 0, stream>>>(y, chStats, chStats + 512, bg, bb, pooled);
        dim3 gl(2, 8);
        linear_wmma_kernel<256><<<gl, 256, 0, stream>>>(pooled, lw, lb, ocat, 256);
        break;
      }
      default: {
        dim3 gc(512 * 4 * 4 / 64, 512 / 64);
        conv_wmma_kernel<512, 4><<<gc, 256, 0, stream>>>(feat, cw, cb, y, chStats, chStats + 512);
        dim3 gp(512, 512);
        bn_pool_kernel<512, 4><<<gp, 128, 0, stream>>>(y, chStats, chStats + 512, bg, bb, pooled);
        dim3 gl(2, 8);
        linear_wmma_kernel<512><<<gl, 256, 0, stream>>>(pooled, lw, lb, ocat, 384);
        break;
      }
    }
  }
  ypred_kernel<<<512, 32, 0, stream>>>(ocat, fin_w, fin_b, out);

  // ----- GTG -----
  e2f16_kernel<<<1024, 256, 0, stream>>>(E, Ef16);
  sq_kernel<<<512, 128, 0, stream>>>(E, sq);
  dim3 gg(8, 8);
  gram_wmma_kernel<<<gg, 256, 0, stream>>>(Ef16, sq, A0);
  seventh_sig_kernel<<<512, 32, 0, stream>>>(A0, E, sig);
  zero_kernel<<<1, 256, 0, stream>>>(Asum, 1);
  a_build_kernel<<<1024, 256, 0, stream>>>(A0, sig, Amat, Asum);
  a_thresh_kernel<<<1024, 256, 0, stream>>>(Amat, Asum);
  x_init_kernel<<<64, 256, 0, stream>>>(labels, X, hist, diff2, flag);
  for (int it = 0; it < MAX_ITER; ++it) {
    gtg_iter_a<<<512, 256, 0, stream>>>(Amat, X, Xn, hist, diff2, flag, it);
    gtg_iter_b<<<1, 256, 0, stream>>>(X, Xn, diff2, flag);
  }
  gtg_finalize<<<20, 256, 0, stream>>>(hist, X, out);
}
